// BaseLayer_67156108640620
// MI455X (gfx1250) — matrix operand order repaired and checked by reading, not compiled, not run on hardware
//
#include <hip/hip_runtime.h>
#include <hip/hip_bf16.h>
#include <stdint.h>

// Problem constants (match reference)
#define S_  2048
#define B_  4
#define D_  1024
#define F_  4096
#define E_  8
#define L_  2
#define T_  (S_*B_)   // 8192 tokens
#define C_  (T_/E_)   // 1024 tokens/expert
#define EPSV 1e-5f

// GEMM tiling
#define BM 128
#define BN 128
#define BK 32
#define LDSR 40       // padded LDS row stride (elements) -> 80B, conflict-free

#define AS1 __attribute__((address_space(1)))
#define AS3 __attribute__((address_space(3)))

#if defined(__gfx1250__) && __has_builtin(__builtin_amdgcn_global_load_async_to_lds_b128) && __has_builtin(__builtin_amdgcn_s_wait_asynccnt)
#define USE_ASYNC_LDS 1
#else
#define USE_ASYNC_LDS 0
#endif

typedef __attribute__((ext_vector_type(16))) __bf16 v16bf;
typedef __attribute__((ext_vector_type(2)))  __bf16 v2bf;
typedef __attribute__((ext_vector_type(8)))  float  v8f;
typedef __attribute__((ext_vector_type(4)))  int    v4i;

union Frag { v16bf v; uint4 q[2]; };

__device__ __forceinline__ unsigned short f2bf(float x) {
  __bf16 b = (__bf16)x;                 // native f32->bf16 (RNE), hw cvt op
  return __builtin_bit_cast(unsigned short, b);
}
// Pack two floats into one DWORD of bf16s -> single v_cvt_pk_bf16_f32
__device__ __forceinline__ unsigned pk_bf16(float lo, float hi) {
  v2bf p;
  p.x = (__bf16)lo;
  p.y = (__bf16)hi;
  return __builtin_bit_cast(unsigned, p);
}

__device__ __forceinline__ float wave_sum(float v) {
  #pragma unroll
  for (int m = 16; m > 0; m >>= 1) v += __shfl_xor(v, m, 32);
  return v;
}

#if USE_ASYNC_LDS
__device__ __forceinline__ void async_copy16(const void* gsrc, void* ldst) {
  // AS1 value == generic value for global memory; AS3 needs a real addrspacecast.
  __builtin_amdgcn_global_load_async_to_lds_b128(
      (AS1 v4i*)(uintptr_t)gsrc,
      (AS3 v4i*)(v4i*)ldst,
      0, 0);
}
#endif

// ---------------------------------------------------------------------------
// K1: affinity = features @ centroids^T  (+ per-token max). Wave per token.
// ---------------------------------------------------------------------------
__global__ __launch_bounds__(256) void affinity_kernel(
    const float* __restrict__ feat, const float* __restrict__ cen,
    float* __restrict__ aff, float* __restrict__ mx) {
  int wid = threadIdx.x >> 5, lane = threadIdx.x & 31;
  int t = blockIdx.x * 8 + wid;
  const float4* f4 = (const float4*)(feat + (size_t)t * D_);
  float acc[E_];
  #pragma unroll
  for (int e = 0; e < E_; ++e) acc[e] = 0.f;
  for (int c = lane; c < D_ / 4; c += 32) {
    float4 x = f4[c];
    #pragma unroll
    for (int e = 0; e < E_; ++e) {
      float4 w = ((const float4*)(cen + (size_t)e * D_))[c];
      acc[e] += x.x * w.x + x.y * w.y + x.z * w.z + x.w * w.w;
    }
  }
  float best = -3.4e38f;
  #pragma unroll
  for (int e = 0; e < E_; ++e) {
    float s = wave_sum(acc[e]);
    if (lane == 0) { aff[(size_t)t * E_ + e] = s; best = fmaxf(best, s); }
  }
  if (lane == 0) mx[t] = best;
}

// ---------------------------------------------------------------------------
// K2: balanced greedy routing. One workgroup: LDS bitonic sort of 8192
// (desc-key, asc-token) packed u64, then serial greedy + stable counting sort.
// ---------------------------------------------------------------------------
__global__ void route_kernel(const float* __restrict__ aff,
                             const float* __restrict__ mx,
                             int* __restrict__ perm) {
  extern __shared__ unsigned long long smem[];
  unsigned long long* key = smem;          // T_ entries
  int* assign = (int*)(smem + T_);         // T_ entries
  int tid = threadIdx.x;

  for (int i = tid; i < T_; i += blockDim.x) {
    unsigned u = __float_as_uint(mx[i]);
    unsigned o = (u & 0x80000000u) ? ~u : (u | 0x80000000u); // orderable asc
    unsigned k = ~o;                                         // descending key
    key[i] = ((unsigned long long)k << 32) | (unsigned)i;
  }
  __syncthreads();

  for (unsigned k2 = 2; k2 <= (unsigned)T_; k2 <<= 1) {
    for (unsigned j = k2 >> 1; j > 0; j >>= 1) {
      for (unsigned i = tid; i < (unsigned)T_; i += blockDim.x) {
        unsigned ixj = i ^ j;
        if (ixj > i) {
          bool up = ((i & k2) == 0);
          unsigned long long a = key[i], b = key[ixj];
          if (up ? (a > b) : (a < b)) { key[i] = b; key[ixj] = a; }
        }
      }
      __syncthreads();
    }
  }

  if (tid == 0) {
    int counts[E_];
    #pragma unroll
    for (int e = 0; e < E_; ++e) counts[e] = 0;
    for (int i = 0; i < T_; ++i) {
      int tok = (int)(key[i] & 0xffffffffu);
      const float* ar = aff + (size_t)tok * E_;
      float bv = -3.4e38f; int be = 0;
      #pragma unroll
      for (int e = 0; e < E_; ++e) {
        float v = (counts[e] < C_) ? ar[e] : -3.4e38f;
        if (v > bv) { bv = v; be = e; }
      }
      counts[be]++; assign[tok] = be;
    }
    int offs[E_];
    #pragma unroll
    for (int e = 0; e < E_; ++e) offs[e] = e * C_;
    for (int t = 0; t < T_; ++t) perm[offs[assign[t]]++] = t;
  }
}

// ---------------------------------------------------------------------------
// K3: gather routed rows + alpha = sigmoid(x . centroid_e). Wave per position.
// ---------------------------------------------------------------------------
__global__ __launch_bounds__(256) void gather_kernel(
    const float* __restrict__ feat, const float* __restrict__ cen,
    const int* __restrict__ perm, float* __restrict__ xr,
    float* __restrict__ alpha) {
  int wid = threadIdx.x >> 5, lane = threadIdx.x & 31;
  int p = blockIdx.x * 8 + wid;
  int tok = perm[p];
  int e = p / C_;
  const float4* src = (const float4*)(feat + (size_t)tok * D_);
  const float4* cw  = (const float4*)(cen  + (size_t)e   * D_);
  float4* dst = (float4*)(xr + (size_t)p * D_);
  float acc = 0.f;
  for (int c = lane; c < D_ / 4; c += 32) {
    float4 x = src[c]; float4 w = cw[c];
    dst[c] = x;
    acc += x.x * w.x + x.y * w.y + x.z * w.z + x.w * w.w;
  }
  acc = wave_sum(acc);
  if (lane == 0) alpha[p] = 1.f / (1.f + expf(-acc));
}

// ---------------------------------------------------------------------------
// K4a: LayerNorm (+affine) -> bf16 activations. Wave per row.
// ---------------------------------------------------------------------------
__global__ __launch_bounds__(256) void ln_kernel(
    const float* __restrict__ xr, const float* __restrict__ gamma,
    const float* __restrict__ beta, unsigned short* __restrict__ hbf, int l) {
  int wid = threadIdx.x >> 5, lane = threadIdx.x & 31;
  int p = blockIdx.x * 8 + wid;
  int e = p / C_;
  const float4* row = (const float4*)(xr + (size_t)p * D_);
  float4 buf[8];
  float s = 0.f, ss = 0.f;
  #pragma unroll
  for (int k = 0; k < 8; ++k) {
    float4 x = row[k * 32 + lane];
    buf[k] = x;
    s  += x.x + x.y + x.z + x.w;
    ss += x.x * x.x + x.y * x.y + x.z * x.z + x.w * x.w;
  }
  s = wave_sum(s); ss = wave_sum(ss);
  float mu  = s * (1.f / D_);
  float var = ss * (1.f / D_) - mu * mu;
  float rs  = rsqrtf(var + EPSV);
  const float4* g4 = (const float4*)(gamma + ((size_t)l * E_ + e) * D_);
  const float4* b4 = (const float4*)(beta  + ((size_t)l * E_ + e) * D_);
  uint2* out = (uint2*)(hbf + (size_t)p * D_);
  #pragma unroll
  for (int k = 0; k < 8; ++k) {
    int c = k * 32 + lane;
    float4 x = buf[k]; float4 g = g4[c]; float4 b = b4[c];
    float hx = (x.x - mu) * rs * g.x + b.x;
    float hy = (x.y - mu) * rs * g.y + b.y;
    float hz = (x.z - mu) * rs * g.z + b.z;
    float hw = (x.w - mu) * rs * g.w + b.w;
    uint2 pk;
    pk.x = pk_bf16(hx, hy);
    pk.y = pk_bf16(hz, hw);
    out[c] = pk;
  }
}

// ---------------------------------------------------------------------------
// WMMA fragment loads (ISA 7.12.2 layouts), from padded LDS tiles
// ---------------------------------------------------------------------------
__device__ __forceinline__ void load_fragA(const unsigned short* tile, int row0,
                                           int lane, Frag& f) {
  int r = lane & 15, hi = lane >> 4;
  const unsigned short* p = tile + (size_t)(row0 + r) * LDSR + hi * 8;
  f.q[0] = *(const uint4*)(p);        // K = hi*8 .. +8
  f.q[1] = *(const uint4*)(p + 16);   // K = 16 + hi*8 .. +8
}
__device__ __forceinline__ void load_fragB(const unsigned short* tile, int col0,
                                           int lane, Frag& f) {
  int n = lane & 15, hi = lane >> 4;
  const unsigned short* p = tile + (size_t)(col0 + n) * LDSR + hi * 8;
  f.q[0] = *(const uint4*)(p);        // K = hi*8 .. +8
  f.q[1] = *(const uint4*)(p + 16);   // K = 16 + hi*8 .. +8
}

// ---------------------------------------------------------------------------
// K4b: H1 = relu(bf16(LN(x)) @ W1^T + b1) -> bf16.  K = D_.
// grid (F_/BN, C_/BM, E_), 256 threads = 8 waves in 2x4.
// ---------------------------------------------------------------------------
__global__ __launch_bounds__(256) void gemm1_kernel(
    const unsigned short* __restrict__ hbf,   // [E_*C_, D_] bf16
    const float* __restrict__ W1,             // [L_,E_,F_,D_] f32
    const float* __restrict__ b1,             // [L_,E_,F_]
    unsigned short* __restrict__ h1,          // [E_*C_, F_] bf16
    int l) {
  __shared__ unsigned short aT[2][BM * LDSR];
  __shared__ unsigned short bT[2][BN * LDSR];

  const int tid = threadIdx.x;
  const int lane = tid & 31, w = tid >> 5;
  const int wm = w >> 2, wn = w & 3;
  const int e  = blockIdx.z;
  const int m0 = blockIdx.y * BM;
  const int n0 = blockIdx.x * BN;

  const unsigned short* Abase = hbf + ((size_t)e * C_ + m0) * D_;
  const float* Bbase = W1 + (((size_t)l * E_ + e) * F_ + n0) * (size_t)D_;

  v8f acc[4][2];
  #pragma unroll
  for (int i = 0; i < 4; ++i)
    #pragma unroll
    for (int j = 0; j < 2; ++j)
      #pragma unroll
      for (int k = 0; k < 8; ++k) acc[i][j][k] = 0.f;

  const int ac0 = tid * 2;  // 512 A 16B-chunks total, 2 per thread
  // ---- stage 0 ----
  #pragma unroll
  for (int j = 0; j < 2; ++j) {
    int ch = ac0 + j, r = ch >> 2, c = ch & 3;
#if USE_ASYNC_LDS
    async_copy16(Abase + (size_t)r * D_ + c * 8, &aT[0][r * LDSR + c * 8]);
#else
    *(uint4*)&aT[0][r * LDSR + c * 8] =
        *(const uint4*)(Abase + (size_t)r * D_ + c * 8);
#endif
  }
  #pragma unroll
  for (int j = 0; j < 4; ++j) {   // 1024 B float4-chunks, 4 per thread
    int ch = tid + 256 * j, r = ch >> 3, c = ch & 7;
    float4 v = *(const float4*)(Bbase + (size_t)r * D_ + c * 4);
    uint2 pk; pk.x = pk_bf16(v.x, v.y); pk.y = pk_bf16(v.z, v.w);
    *(uint2*)&bT[0][r * LDSR + c * 4] = pk;
  }
#if USE_ASYNC_LDS
  __builtin_amdgcn_s_wait_asynccnt(0);
#endif
  __syncthreads();

  const int KT = D_ / BK;
#if !USE_ASYNC_LDS
  uint4 aReg[2];
#endif
  float4 bReg[4];
  for (int kt = 0; kt < KT; ++kt) {
    int cur = kt & 1, nxt = cur ^ 1;
    int k0n = (kt + 1) * BK;
    if (kt + 1 < KT) {
      #pragma unroll
      for (int j = 0; j < 2; ++j) {
        int ch = ac0 + j, r = ch >> 2, c = ch & 3;
#if USE_ASYNC_LDS
        async_copy16(Abase + (size_t)r * D_ + k0n + c * 8,
                     &aT[nxt][r * LDSR + c * 8]);
#else
        aReg[j] = *(const uint4*)(Abase + (size_t)r * D_ + k0n + c * 8);
#endif
      }
      #pragma unroll
      for (int j = 0; j < 4; ++j) {
        int ch = tid + 256 * j, r = ch >> 3, c = ch & 7;
        bReg[j] = *(const float4*)(Bbase + (size_t)r * D_ + k0n + c * 4);
      }
      if (kt + 2 < KT)  // CDNA5 global_prefetch_b8 for tile after next
        __builtin_prefetch(Bbase + (size_t)(tid >> 3) * D_ + (kt + 2) * BK, 0, 1);
    }

    Frag a[4], b[2];
    #pragma unroll
    for (int i = 0; i < 4; ++i) load_fragA(aT[cur], wm * 64 + i * 16, lane, a[i]);
    #pragma unroll
    for (int j = 0; j < 2; ++j) load_fragB(bT[cur], wn * 32 + j * 16, lane, b[j]);
    #pragma unroll
    for (int i = 0; i < 4; ++i)
      #pragma unroll
      for (int j = 0; j < 2; ++j)
        acc[i][j] = __builtin_amdgcn_wmma_f32_16x16x32_bf16(
            false, a[i].v, false, b[j].v, (short)0, acc[i][j], false, false);

    if (kt + 1 < KT) {
#if !USE_ASYNC_LDS
      #pragma unroll
      for (int j = 0; j < 2; ++j) {
        int ch = ac0 + j, r = ch >> 2, c = ch & 3;
        *(uint4*)&aT[nxt][r * LDSR + c * 8] = aReg[j];
      }
#endif
      #pragma unroll
      for (int j = 0; j < 4; ++j) {
        int ch = tid + 256 * j, r = ch >> 3, c = ch & 7;
        uint2 pk; pk.x = pk_bf16(bReg[j].x, bReg[j].y);
        pk.y = pk_bf16(bReg[j].z, bReg[j].w);
        *(uint2*)&bT[nxt][r * LDSR + c * 4] = pk;
      }
#if USE_ASYNC_LDS
      __builtin_amdgcn_s_wait_asynccnt(0);
#endif
    }
    __syncthreads();
  }

  // epilogue: bias + relu + bf16 store
  #pragma unroll
  for (int j = 0; j < 2; ++j) {
    int col = n0 + wn * 32 + j * 16 + (lane & 15);
    float bias = b1[((size_t)l * E_ + e) * F_ + col];
    #pragma unroll
    for (int i = 0; i < 4; ++i) {
      #pragma unroll
      for (int vk = 0; vk < 8; ++vk) {
        int row = m0 + wm * 64 + i * 16 + vk + ((lane >> 4) * 8);
        float vv = fmaxf(acc[i][j][vk] + bias, 0.f);
        h1[((size_t)e * C_ + row) * F_ + col] = f2bf(vv);
      }
    }
  }
}

// ---------------------------------------------------------------------------
// K4c: x += H1 @ W2^T + b2.  K = F_.  grid (D_/BN, C_/BM, E_).
// ---------------------------------------------------------------------------
__global__ __launch_bounds__(256) void gemm2_kernel(
    const unsigned short* __restrict__ h1,    // [E_*C_, F_] bf16
    const float* __restrict__ W2,             // [L_,E_,D_,F_] f32
    const float* __restrict__ b2,             // [L_,E_,D_]
    float* __restrict__ xr,                   // [E_*C_, D_] f32 (residual +=)
    int l) {
  __shared__ unsigned short aT[2][BM * LDSR];
  __shared__ unsigned short bT[2][BN * LDSR];

  const int tid = threadIdx.x;
  const int lane = tid & 31, w = tid >> 5;
  const int wm = w >> 2, wn = w & 3;
  const int e  = blockIdx.z;
  const int m0 = blockIdx.y * BM;
  const int n0 = blockIdx.x * BN;

  const unsigned short* Abase = h1 + ((size_t)e * C_ + m0) * F_;
  const float* Bbase = W2 + (((size_t)l * E_ + e) * D_ + n0) * (size_t)F_;

  v8f acc[4][2];
  #pragma unroll
  for (int i = 0; i < 4; ++i)
    #pragma unroll
    for (int j = 0; j < 2; ++j)
      #pragma unroll
      for (int k = 0; k < 8; ++k) acc[i][j][k] = 0.f;

  const int ac0 = tid * 2;
  #pragma unroll
  for (int j = 0; j < 2; ++j) {
    int ch = ac0 + j, r = ch >> 2, c = ch & 3;
#if USE_ASYNC_LDS
    async_copy16(Abase + (size_t)r * F_ + c * 8, &aT[0][r * LDSR + c * 8]);
#else
    *(uint4*)&aT[0][r * LDSR + c * 8] =
        *(const uint4*)(Abase + (size_t)r * F_ + c * 8);
#endif
  }
  #pragma unroll
  for (int j = 0; j < 4; ++j) {
    int ch = tid + 256 * j, r = ch >> 3, c = ch & 7;
    float4 v = *(const float4*)(Bbase + (size_t)r * F_ + c * 4);
    uint2 pk; pk.x = pk_bf16(v.x, v.y); pk.y = pk_bf16(v.z, v.w);
    *(uint2*)&bT[0][r * LDSR + c * 4] = pk;
  }
#if USE_ASYNC_LDS
  __builtin_amdgcn_s_wait_asynccnt(0);
#endif
  __syncthreads();

  const int KT = F_ / BK;
#if !USE_ASYNC_LDS
  uint4 aReg[2];
#endif
  float4 bReg[4];
  for (int kt = 0; kt < KT; ++kt) {
    int cur = kt & 1, nxt = cur ^ 1;
    int k0n = (kt + 1) * BK;
    if (kt + 1 < KT) {
      #pragma unroll
      for (int j = 0; j < 2; ++j) {
        int ch = ac0 + j, r = ch >> 2, c = ch & 3;
#if USE_ASYNC_LDS
        async_copy16(Abase + (size_t)r * F_ + k0n + c * 8,
                     &aT[nxt][r * LDSR + c * 8]);
#else
        aReg[j] = *(const uint4*)(Abase + (size_t)r * F_ + k0n + c * 8);
#endif
      }
      #pragma unroll
      for (int j = 0; j < 4; ++j) {
        int ch = tid + 256 * j, r = ch >> 3, c = ch & 7;
        bReg[j] = *(const float4*)(Bbase + (size_t)r * F_ + k0n + c * 4);
      }
      if (kt + 2 < KT)
        __builtin_prefetch(Bbase + (size_t)(tid >> 3) * F_ + (kt + 2) * BK, 0, 1);
    }

    Frag a[4], b[2];
    #pragma unroll
    for (int i = 0; i < 4; ++i) load_fragA(aT[cur], wm * 64 + i * 16, lane, a[i]);
    #pragma unroll
    for (int j = 0; j < 2; ++j) load_fragB(bT[cur], wn * 32 + j * 16, lane, b[j]);
    #pragma unroll
    for (int i = 0; i < 4; ++i)
      #pragma unroll
      for (int j = 0; j < 2; ++j)
        acc[i][j] = __builtin_amdgcn_wmma_f32_16x16x32_bf16(
            false, a[i].v, false, b[j].v, (short)0, acc[i][j], false, false);

    if (kt + 1 < KT) {
#if !USE_ASYNC_LDS
      #pragma unroll
      for (int j = 0; j < 2; ++j) {
        int ch = ac0 + j, r = ch >> 2, c = ch & 3;
        *(uint4*)&aT[nxt][r * LDSR + c * 8] = aReg[j];
      }
#endif
      #pragma unroll
      for (int j = 0; j < 4; ++j) {
        int ch = tid + 256 * j, r = ch >> 3, c = ch & 7;
        uint2 pk; pk.x = pk_bf16(bReg[j].x, bReg[j].y);
        pk.y = pk_bf16(bReg[j].z, bReg[j].w);
        *(uint2*)&bT[nxt][r * LDSR + c * 4] = pk;
      }
#if USE_ASYNC_LDS
      __builtin_amdgcn_s_wait_asynccnt(0);
#endif
    }
    __syncthreads();
  }

  // epilogue: residual add + bias
  #pragma unroll
  for (int j = 0; j < 2; ++j) {
    int col = n0 + wn * 32 + j * 16 + (lane & 15);
    float bias = b2[((size_t)l * E_ + e) * D_ + col];
    #pragma unroll
    for (int i = 0; i < 4; ++i) {
      #pragma unroll
      for (int vk = 0; vk < 8; ++vk) {
        int row = m0 + wm * 64 + i * 16 + vk + ((lane >> 4) * 8);
        float* xp = &xr[((size_t)e * C_ + row) * D_ + col];
        *xp = *xp + acc[i][j][vk] + bias;
      }
    }
  }
}

// ---------------------------------------------------------------------------
// K5: out[token] = alpha[p] * x[p]  (perm is a full permutation -> no zeroing)
// ---------------------------------------------------------------------------
__global__ __launch_bounds__(256) void scatter_kernel(
    const float* __restrict__ xr, const float* __restrict__ alpha,
    const int* __restrict__ perm, float* __restrict__ out) {
  int wid = threadIdx.x >> 5, lane = threadIdx.x & 31;
  int p = blockIdx.x * 8 + wid;
  int tok = perm[p];
  float a = alpha[p];
  const float4* src = (const float4*)(xr + (size_t)p * D_);
  float4* dst = (float4*)(out + (size_t)tok * D_);
  for (int c = lane; c < D_ / 4; c += 32) {
    float4 x = src[c];
    x.x *= a; x.y *= a; x.z *= a; x.w *= a;
    dst[c] = x;
  }
}

// ---------------------------------------------------------------------------
extern "C" void kernel_launch(void* const* d_in, const int* in_sizes, int n_in,
                              void* d_out, int out_size, void* d_ws, size_t ws_size,
                              hipStream_t stream) {
  const float* feat = (const float*)d_in[0];   // [S,B,D]
  const float* cen  = (const float*)d_in[1];   // [E,D]
  const float* gam  = (const float*)d_in[2];   // [L,E,D]
  const float* bet  = (const float*)d_in[3];   // [L,E,D]
  const float* W1   = (const float*)d_in[4];   // [L,E,F,D]
  const float* b1   = (const float*)d_in[5];   // [L,E,F]
  const float* W2   = (const float*)d_in[6];   // [L,E,D,F]
  const float* b2   = (const float*)d_in[7];   // [L,E,D]
  float* out = (float*)d_out;
  (void)in_sizes; (void)n_in; (void)out_size; (void)ws_size;

  char* ws = (char*)d_ws;
  size_t off = 0;
  auto wsalloc = [&](size_t bytes) -> void* {
    off = (off + 255) & ~(size_t)255;
    void* p = ws + off;
    off += bytes;
    return p;
  };
  float* aff   = (float*)wsalloc((size_t)T_ * E_ * 4);
  float* mx    = (float*)wsalloc((size_t)T_ * 4);
  int*   perm  = (int*)  wsalloc((size_t)T_ * 4);
  float* alpha = (float*)wsalloc((size_t)T_ * 4);
  float* xr    = (float*)wsalloc((size_t)T_ * D_ * 4);
  unsigned short* hbf = (unsigned short*)wsalloc((size_t)T_ * D_ * 2);
  unsigned short* h1  = (unsigned short*)wsalloc((size_t)T_ * F_ * 2);

  affinity_kernel<<<T_ / 8, 256, 0, stream>>>(feat, cen, aff, mx);
  route_kernel<<<1, 1024, (size_t)T_ * 8 + (size_t)T_ * 4, stream>>>(aff, mx, perm);
  gather_kernel<<<T_ / 8, 256, 0, stream>>>(feat, cen, perm, xr, alpha);
  for (int l = 0; l < L_; ++l) {
    ln_kernel<<<T_ / 8, 256, 0, stream>>>(xr, gam, bet, hbf, l);
    gemm1_kernel<<<dim3(F_ / BN, C_ / BM, E_), 256, 0, stream>>>(hbf, W1, b1, h1, l);
    gemm2_kernel<<<dim3(D_ / BN, C_ / BM, E_), 256, 0, stream>>>(h1, W2, b2, xr, l);
  }
  scatter_kernel<<<T_ / 8, 256, 0, stream>>>(xr, alpha, perm, out);
}
